// TFNLayer_75531294867757
// MI455X (gfx1250) — matrix operand order repaired and checked by reading, not compiled, not run on hardware
//
#include <hip/hip_runtime.h>
#include <stdint.h>

// ---------------------------------------------------------------------------
// TFN layer for MI455X (gfx1250, wave32, WMMA 16x16x32 f16).
//
// N=512 nodes, per-edge MLPs 16->32->256 folded with per-node features into
// per-j 32x16 f16 matrices (8 per j, 8KB/j, 4MB total -> lives in 192MB L2).
//   msg = sum_j ( scale(i,j) * silu(wa@rbf+ba) ) x B[j]
// Stage-1 is computed transposed (A=wa const, B=rbf tile) so its f32 C/D
// layout coincides exactly with the f16 A-operand layout of stage-2:
//   C/D:  lane L -> N=i=L%16, M=e=v+8*(L>>4)
//   A16:  lane L -> M=L%16,  K-set {0..7,16..23} (L<16) / {8..15,24..31},
//         slot = (K&7) + 8*(K>>4)
// => silu + mask/r_hat scaling + cvt f16 entirely in VGPRs, no LDS transpose.
//
// Round-2 changes: fast v_rcp sigmoid (kill IEEE div chains), packed-f16
// scaling of the r_hat-weighted A variants (v_pk_mul_f16 instead of f32
// mul + cvt per variant).
// ---------------------------------------------------------------------------

#define NN   512
#define EPSF 1e-5f

typedef __attribute__((ext_vector_type(16))) _Float16 v16h;
typedef __attribute__((ext_vector_type(8)))  float    v8f;
typedef __attribute__((ext_vector_type(4)))  float    v4f;

static __device__ __forceinline__ v8f wmma_f16(v16h a, v16h b, v8f c) {
  // D = A(16x32 f16) x B(32x16 f16) + C(16x16 f32)
  return __builtin_amdgcn_wmma_f32_16x16x32_f16(false, a, false, b, (short)0, c,
                                                false, false);
}

static __device__ __forceinline__ float silu(float x) {
  // x * sigmoid(x); denominator >= 1 so fast v_rcp_f32 is safe.
  return x * __builtin_amdgcn_rcpf(1.0f + __expf(-x));
}

// masked silu of 16 hidden values (two stage-1 WMMA halves) -> packed f16 A.
static __device__ __forceinline__ v16h silu_mask_cvt(v8f h0, v8f h1, float mk) {
  v16h a;
  #pragma unroll
  for (int r = 0; r < 8; ++r) {
    a[r]     = (_Float16)(mk * silu(h0[r]));
    a[r + 8] = (_Float16)(mk * silu(h1[r]));
  }
  return a;
}

// ---------------------------------------------------------------------------
// Prep kernel: fold wb/bb with per-node features into per-j B-matrices
// (stored pre-swizzled for the WMMA 16-bit B operand: lane = o + 16*((h>>3)&1),
// slot = (h&7) + 8*(h>>4)) and per-j bias vectors c (f32).
//   tile 0: B00[h][o] = sum_d f0[j,d]   * wb00[(d*16+o),h]
//   tile 1..3 (k): B10k = sum_c f1[j,c,k] * wb10[(c*16+o),h]
//   tile 4: B01  = sum_f f0[j,f]        * wb01[(f*16+o),h]
//   tile 5..7 (x): B11x = sum_k f1[j,k,x]* wb11[(k*16+o),h]
// ---------------------------------------------------------------------------
__global__ __launch_bounds__(256) void tfn_prep(
    const float* __restrict__ f0, const float* __restrict__ f1,
    const float* __restrict__ w00b, const float* __restrict__ b00b,
    const float* __restrict__ w10b, const float* __restrict__ b10b,
    const float* __restrict__ w01b, const float* __restrict__ b01b,
    const float* __restrict__ w11b, const float* __restrict__ b11b,
    _Float16* __restrict__ wsB, float* __restrict__ wsC)
{
  const int j = blockIdx.x;
  const int t = threadIdx.x;

  __shared__ float s_f0[16];
  __shared__ float s_f1[48];
  if (t < 16) s_f0[t] = f0[j * 16 + t];
  if (t < 48) s_f1[t] = f1[j * 48 + t];
  __syncthreads();

  // 8 tiles * 512 entries = 4096 entries per j; 16 per thread.
  for (int q = 0; q < 16; ++q) {
    const int idx = t + 256 * q;       // 0..4095
    const int m   = idx >> 9;          // tile 0..7
    const int r   = idx & 511;
    const int h   = r >> 4;            // hidden 0..31
    const int o   = r & 15;            // out col 0..15
    float acc = 0.0f;
    if (m == 0) {
      for (int d = 0; d < 16; ++d) acc += s_f0[d] * w00b[(d * 16 + o) * 32 + h];
    } else if (m < 4) {
      const int k = m - 1;
      for (int c = 0; c < 16; ++c) acc += s_f1[c * 3 + k] * w10b[(c * 16 + o) * 32 + h];
    } else if (m == 4) {
      for (int f = 0; f < 16; ++f) acc += s_f0[f] * w01b[(f * 16 + o) * 32 + h];
    } else {
      const int x = m - 5;
      for (int k = 0; k < 16; ++k) acc += s_f1[k * 3 + x] * w11b[(k * 16 + o) * 32 + h];
    }
    const int lane = o + 16 * ((h >> 3) & 1);
    const int slot = (h & 7) + 8 * (h >> 4);
    wsB[(size_t)j * 4096 + m * 512 + lane * 16 + slot] = (_Float16)acc;
  }

  // per-j bias vectors: 8 sets of 16 floats
  if (t < 128) {
    const int m = t >> 4, o = t & 15;
    float acc = 0.0f;
    if (m == 0) {
      for (int d = 0; d < 16; ++d) acc += s_f0[d] * b00b[d * 16 + o];
    } else if (m < 4) {
      const int k = m - 1;
      for (int c = 0; c < 16; ++c) acc += s_f1[c * 3 + k] * b10b[c * 16 + o];
    } else if (m == 4) {
      for (int f = 0; f < 16; ++f) acc += s_f0[f] * b01b[f * 16 + o];
    } else {
      const int x = m - 5;
      for (int k = 0; k < 16; ++k) acc += s_f1[k * 3 + x] * b11b[k * 16 + o];
    }
    wsC[j * 128 + t] = acc;
  }
}

// ---------------------------------------------------------------------------
// Main kernel: per wave one i-block of 16 rows, loop over its j slice,
// 18 v_wmma_f32_16x16x32_f16 per (i16, j): 8 stage-1 + 10 stage-2.
// Grid (32 i-blocks, 8 j-chunks), 256 threads = 8 waves, 8 j per wave.
// ---------------------------------------------------------------------------
__global__ __launch_bounds__(256) void tfn_main(
    const float* __restrict__ rbf, const float* __restrict__ r_hat,
    const uint8_t* __restrict__ mask,
    const float* __restrict__ w00a, const float* __restrict__ b00a,
    const float* __restrict__ w10a, const float* __restrict__ b10a,
    const float* __restrict__ w01a, const float* __restrict__ b01a,
    const float* __restrict__ w11a, const float* __restrict__ b11a,
    const _Float16* __restrict__ wsB,
    float* __restrict__ msg_s, float* __restrict__ msg_v)
{
  const int lane = threadIdx.x & 31;
  const int wv   = threadIdx.x >> 5;
  const int hi   = lane >> 4;     // 0/1 half-wave
  const int il   = lane & 15;
  const int iblk = blockIdx.x << 4;
  const int irow = iblk + il;

  const float* waP[4] = { w00a, w10a, w01a, w11a };
  const float* baP[4] = { b00a, b10a, b01a, b11a };

  // Stage-1 constant A operands (wa, K=rbf padded 16->32) and bias-as-C.
  v16h awa[4][2];
  v8f  bav[4][2];
  #pragma unroll
  for (int m = 0; m < 4; ++m) {
    #pragma unroll
    for (int c = 0; c < 2; ++c) {
      const float* wp = waP[m] + (c * 16 + il) * 16 + hi * 8;
      v16h a;
      #pragma unroll
      for (int r = 0; r < 8; ++r) a[r] = (_Float16)wp[r];
      #pragma unroll
      for (int r = 8; r < 16; ++r) a[r] = (_Float16)0.0f;  // K=16..31 pad
      awa[m][c] = a;
      const float* bp = baP[m] + c * 16 + hi * 8;
      v8f b;
      #pragma unroll
      for (int r = 0; r < 8; ++r) b[r] = bp[r];
      bav[m][c] = b;
    }
  }

  v8f accS  = {};                 // new_f0 pre-LN (msg00+msg10)
  v8f accV0 = {}, accV1 = {}, accV2 = {};  // new_f1_raw x-components

  const int j0 = blockIdx.y * 64 + wv * 8;
  for (int jj = 0; jj < 8; ++jj) {
    const int j = j0 + jj;
    const _Float16* bj = wsB + (size_t)j * 4096;
    if (jj + 1 < 8)  // pull next j's folded-B block (L2-resident) closer
      __builtin_prefetch(wsB + (size_t)(j + 1) * 4096 + lane * 8, 0, 1);

    // Stage-1 B operand: rbf row (16 f32 contiguous) -> f16.
    // Elements 8..15 of each lane carry K=16..31 which multiplies zero A columns, so the
    // duplicated (finite) values there are mathematically inert.
    const float* rp = rbf + ((size_t)irow * NN + j) * 16;
    v16h brbf;
    #pragma unroll
    for (int s = 0; s < 16; s += 4) {
      v4f tv = *(const v4f*)(rp + ((hi * 8 + s) & 15));
      brbf[s + 0] = (_Float16)tv.x;
      brbf[s + 1] = (_Float16)tv.y;
      brbf[s + 2] = (_Float16)tv.z;
      brbf[s + 3] = (_Float16)tv.w;
    }

    const float mk = mask[(size_t)irow * NN + j] ? 1.0f : 0.0f;
    const float* rh = r_hat + ((size_t)irow * NN + j) * 3;
    const _Float16 rv[3] = { (_Float16)rh[0], (_Float16)rh[1], (_Float16)rh[2] };

    // ---- MLP 00 : scalar -> scalar ----
    {
      v8f h0 = wmma_f16(awa[0][0], brbf, bav[0][0]);
      v8f h1 = wmma_f16(awa[0][1], brbf, bav[0][1]);
      v16h A = silu_mask_cvt(h0, h1, mk);
      v16h B = *(const v16h*)(bj + 0 * 512 + lane * 16);
      accS = wmma_f16(A, B, accS);
    }
    // ---- MLP 10 : vector -> scalar (fold r_hat into A rows, pk_mul f16) ----
    {
      v8f h0 = wmma_f16(awa[1][0], brbf, bav[1][0]);
      v8f h1 = wmma_f16(awa[1][1], brbf, bav[1][1]);
      v16h base = silu_mask_cvt(h0, h1, mk);
      #pragma unroll
      for (int k = 0; k < 3; ++k) {
        v16h A = base * rv[k];   // v_pk_mul_f16 x8 (scalar splat)
        v16h B = *(const v16h*)(bj + (1 + k) * 512 + lane * 16);
        accS = wmma_f16(A, B, accS);
      }
    }
    // ---- MLP 01 : scalar -> vector (same B, 3 scaled A) ----
    {
      v8f h0 = wmma_f16(awa[2][0], brbf, bav[2][0]);
      v8f h1 = wmma_f16(awa[2][1], brbf, bav[2][1]);
      v16h base = silu_mask_cvt(h0, h1, mk);
      v16h B = *(const v16h*)(bj + 4 * 512 + lane * 16);
      accV0 = wmma_f16(base * rv[0], B, accV0);
      accV1 = wmma_f16(base * rv[1], B, accV1);
      accV2 = wmma_f16(base * rv[2], B, accV2);
    }
    // ---- MLP 11 : vector -> vector (same A, 3 B tiles) ----
    {
      v8f h0 = wmma_f16(awa[3][0], brbf, bav[3][0]);
      v8f h1 = wmma_f16(awa[3][1], brbf, bav[3][1]);
      v16h A = silu_mask_cvt(h0, h1, mk);
      accV0 = wmma_f16(A, *(const v16h*)(bj + 5 * 512 + lane * 16), accV0);
      accV1 = wmma_f16(A, *(const v16h*)(bj + 6 * 512 + lane * 16), accV1);
      accV2 = wmma_f16(A, *(const v16h*)(bj + 7 * 512 + lane * 16), accV2);
    }
  }

  // C/D layout: lane -> (N=il, M=v+8*hi). Accumulate partial j-sums.
  const int ib = iblk + hi * 8;
  #pragma unroll
  for (int v = 0; v < 8; ++v) {
    const int row = ib + v;
    atomicAdd(&msg_s[row * 16 + il],               accS[v]);
    atomicAdd(&msg_v[0 * NN * 16 + row * 16 + il], accV0[v]);
    atomicAdd(&msg_v[1 * NN * 16 + row * 16 + il], accV1[v]);
    atomicAdd(&msg_v[2 * NN * 16 + row * 16 + il], accV2[v]);
  }
}

// ---------------------------------------------------------------------------
// Bias kernel: second-layer bias contributions (84 MFLOP, VALU).
// ---------------------------------------------------------------------------
__global__ __launch_bounds__(64) void tfn_bias(
    const uint8_t* __restrict__ mask, const float* __restrict__ r_hat,
    const float* __restrict__ wsC,
    float* __restrict__ msg_s, float* __restrict__ msg_v)
{
  const int i = blockIdx.x;
  const int t = threadIdx.x;      // 0..63
  const int setid = t >> 4;       // 0: scalar, 1..3: vector x-component
  const int o = t & 15;
  float acc = 0.0f;
  for (int j = 0; j < NN; ++j) {
    if (!mask[(size_t)i * NN + j]) continue;   // uniform branch per block
    const float* c  = wsC + j * 128;
    const float* rh = r_hat + ((size_t)i * NN + j) * 3;
    if (setid == 0) {
      acc += c[o] + rh[0] * c[16 + o] + rh[1] * c[32 + o] + rh[2] * c[48 + o];
    } else {
      const int x = setid - 1;
      acc += rh[x] * c[64 + o] + c[80 + x * 16 + o];
    }
  }
  if (setid == 0) msg_s[i * 16 + o] += acc;                       // unique writer
  else            msg_v[(setid - 1) * NN * 16 + i * 16 + o] += acc;
}

// ---------------------------------------------------------------------------
// Finalize: layernorm of new_f0; vector-norm + layernorm-of-norms for new_f1.
// ---------------------------------------------------------------------------
__global__ __launch_bounds__(32) void tfn_final(
    const float* __restrict__ msg_s, const float* __restrict__ msg_v,
    const float* __restrict__ g0, const float* __restrict__ be0,
    const float* __restrict__ g1, const float* __restrict__ be1,
    float* __restrict__ out)
{
  const int i = blockIdx.x;
  const int t = threadIdx.x;

  float xs[16];
  #pragma unroll
  for (int o = 0; o < 16; ++o) xs[o] = msg_s[i * 16 + o];
  float mu = 0.0f;
  #pragma unroll
  for (int o = 0; o < 16; ++o) mu += xs[o];
  mu *= (1.0f / 16.0f);
  float var = 0.0f;
  #pragma unroll
  for (int o = 0; o < 16; ++o) { float d = xs[o] - mu; var += d * d; }
  var *= (1.0f / 16.0f);
  const float inv = rsqrtf(var + EPSF);
  if (t < 16) out[i * 16 + t] = (xs[t] - mu) * inv * g0[t] + be0[t];

  float nr[16];
  #pragma unroll
  for (int g = 0; g < 16; ++g) {
    const float a = msg_v[0 * NN * 16 + i * 16 + g];
    const float b = msg_v[1 * NN * 16 + i * 16 + g];
    const float c = msg_v[2 * NN * 16 + i * 16 + g];
    nr[g] = fmaxf(sqrtf(a * a + b * b + c * c), 1e-8f);
  }
  float mu2 = 0.0f;
  #pragma unroll
  for (int g = 0; g < 16; ++g) mu2 += nr[g];
  mu2 *= (1.0f / 16.0f);
  float var2 = 0.0f;
  #pragma unroll
  for (int g = 0; g < 16; ++g) { float d = nr[g] - mu2; var2 += d * d; }
  var2 *= (1.0f / 16.0f);
  const float inv2 = rsqrtf(var2 + EPSF);
  if (t < 16) {
    const int g = t;
    const float ln = (nr[g] - mu2) * inv2 * g1[g] + be1[g];
    const float s  = ln / nr[g];
    out[NN * 16 + i * 48 + g * 3 + 0] = msg_v[0 * NN * 16 + i * 16 + g] * s;
    out[NN * 16 + i * 48 + g * 3 + 1] = msg_v[1 * NN * 16 + i * 16 + g] * s;
    out[NN * 16 + i * 48 + g * 3 + 2] = msg_v[2 * NN * 16 + i * 16 + g] * s;
  }
}

// ---------------------------------------------------------------------------
extern "C" void kernel_launch(void* const* d_in, const int* in_sizes, int n_in,
                              void* d_out, int out_size, void* d_ws, size_t ws_size,
                              hipStream_t stream) {
  (void)in_sizes; (void)n_in; (void)out_size; (void)ws_size;
  // input order per setup_inputs(): pos,f0,f1,rbf,r_hat,mask, then
  // {w00a,b00a,w00b,b00b},{w10..},{w01..},{w11..}, g0,be0,g1,be1
  const float*   f0    = (const float*)d_in[1];
  const float*   f1    = (const float*)d_in[2];
  const float*   rbf   = (const float*)d_in[3];
  const float*   r_hat = (const float*)d_in[4];
  const uint8_t* mask  = (const uint8_t*)d_in[5];
  const float* w00a = (const float*)d_in[6],  *b00a = (const float*)d_in[7];
  const float* w00b = (const float*)d_in[8],  *b00b = (const float*)d_in[9];
  const float* w10a = (const float*)d_in[10], *b10a = (const float*)d_in[11];
  const float* w10b = (const float*)d_in[12], *b10b = (const float*)d_in[13];
  const float* w01a = (const float*)d_in[14], *b01a = (const float*)d_in[15];
  const float* w01b = (const float*)d_in[16], *b01b = (const float*)d_in[17];
  const float* w11a = (const float*)d_in[18], *b11a = (const float*)d_in[19];
  const float* w11b = (const float*)d_in[20], *b11b = (const float*)d_in[21];
  const float* g0  = (const float*)d_in[22], *be0 = (const float*)d_in[23];
  const float* g1  = (const float*)d_in[24], *be1 = (const float*)d_in[25];
  float* out = (float*)d_out;

  // workspace layout: [0,4MB) folded B f16; [4MB,+256KB) bias c f32;
  // then msg_s (32KB) and msg_v (96KB) f32 accumulators.
  char* ws = (char*)d_ws;
  _Float16* wsB = (_Float16*)ws;
  float* wsC   = (float*)(ws + (4u << 20));
  float* msg_s = (float*)(ws + (4u << 20) + (256u << 10));
  float* msg_v = msg_s + NN * 16;

  hipMemsetAsync(msg_s, 0, (size_t)(NN * 16 + 3 * NN * 16) * sizeof(float), stream);
  tfn_prep<<<NN, 256, 0, stream>>>(f0, f1, w00b, b00b, w10b, b10b,
                                   w01b, b01b, w11b, b11b, wsB, wsC);
  tfn_main<<<dim3(NN / 16, 8), 256, 0, stream>>>(rbf, r_hat, mask,
                                                 w00a, b00a, w10a, b10a,
                                                 w01a, b01a, w11a, b11a,
                                                 wsB, msg_s, msg_v);
  tfn_bias<<<NN, 64, 0, stream>>>(mask, r_hat, wsC, msg_s, msg_v);
  tfn_final<<<NN, 32, 0, stream>>>(msg_s, msg_v, g0, be0, g1, be1, out);
}
